// MagiMLP_49263274885543
// MI455X (gfx1250) — matrix operand order repaired and checked by reading, not compiled, not run on hardware
//
#include <hip/hip_runtime.h>
#include <hip/hip_bf16.h>
#include <stdint.h>

// ---- problem constants (from reference) ----
#define N_TOK   8192
#define H_DIM   2048
#define INTER_D 8192
#define M_MOD   2
#define UPOUT   (2 * INTER_D)
#define G_TOK   (N_TOK / M_MOD)     // 4096 tokens per modality
#define EPSV    1e-6f
#define ALPHA_V 1.702f
#define LIMIT_V 7.0f

typedef __bf16 bf16;
typedef __attribute__((ext_vector_type(16))) __bf16 v16bf;
typedef __attribute__((ext_vector_type(8)))  float  v8f;
typedef __attribute__((ext_vector_type(4)))  __bf16 bf16x4;
typedef __attribute__((ext_vector_type(4)))  int    v4i;

// ---- CDNA5 async global->LDS copy (ASYNCcnt path), with safe fallback ----
#if __has_builtin(__builtin_amdgcn_global_load_async_to_lds_b128)
#define HAVE_ASYNC_LDS 1
typedef __attribute__((address_space(1))) v4i* as1_v4i;   // prints as "int4 __device__*"
typedef __attribute__((address_space(3))) v4i* as3_v4i;   // prints as "int4 __shared__*"
#else
#define HAVE_ASYNC_LDS 0
#endif

__device__ __forceinline__ void lds_async_b128(const void* g, void* l) {
#if HAVE_ASYNC_LDS
    // generic->AS1 via 64-bit int; generic LDS ptr low 32 bits == LDS offset (AS3)
    __builtin_amdgcn_global_load_async_to_lds_b128(
        (as1_v4i)(uintptr_t)g, (as3_v4i)(uint32_t)(uintptr_t)l, 0, 0);
#else
    (void)g; (void)l;
#endif
}
__device__ __forceinline__ void wait_async0() {
#if HAVE_ASYNC_LDS
#if __has_builtin(__builtin_amdgcn_s_wait_asynccnt)
    __builtin_amdgcn_s_wait_asynccnt(0);
#else
    asm volatile("s_wait_asynccnt 0x0" ::: "memory");
#endif
#endif
}

// =====================================================================
// Kernel 1: fused MultiModalityRMSNorm -> bf16
// =====================================================================
__global__ __launch_bounds__(256) void magi_rmsnorm_kernel(
    const float* __restrict__ x, const float* __restrict__ w_norm,
    bf16* __restrict__ tm)
{
    __shared__ float red[8];
    const int row = blockIdx.x;
    const int mod = row / G_TOK;                // pre-grouped contiguous halves
    const int t   = threadIdx.x;
    const float4* xr = (const float4*)(x + (size_t)row * H_DIM);
    const float4* wr = (const float4*)(w_norm + (size_t)mod * H_DIM);

    float4 v0 = xr[t];
    float4 v1 = xr[t + 256];
    float ss = v0.x*v0.x + v0.y*v0.y + v0.z*v0.z + v0.w*v0.w
             + v1.x*v1.x + v1.y*v1.y + v1.z*v1.z + v1.w*v1.w;
    #pragma unroll
    for (int off = 16; off > 0; off >>= 1) ss += __shfl_down(ss, off, 32);
    if ((t & 31) == 0) red[t >> 5] = ss;
    __syncthreads();
    float tot = 0.f;
    #pragma unroll
    for (int i = 0; i < 8; ++i) tot += red[i];
    const float rinv = rsqrtf(tot * (1.0f / H_DIM) + EPSV);

    float4 w0 = wr[t];
    float4 w1 = wr[t + 256];
    bf16x4 o0, o1;
    o0[0] = (bf16)(v0.x * rinv * (w0.x + 1.0f));
    o0[1] = (bf16)(v0.y * rinv * (w0.y + 1.0f));
    o0[2] = (bf16)(v0.z * rinv * (w0.z + 1.0f));
    o0[3] = (bf16)(v0.w * rinv * (w0.w + 1.0f));
    o1[0] = (bf16)(v1.x * rinv * (w1.x + 1.0f));
    o1[1] = (bf16)(v1.y * rinv * (w1.y + 1.0f));
    o1[2] = (bf16)(v1.z * rinv * (w1.z + 1.0f));
    o1[3] = (bf16)(v1.w * rinv * (w1.w + 1.0f));
    bf16x4* out = (bf16x4*)(tm + (size_t)row * H_DIM);
    out[t]       = o0;
    out[t + 256] = o1;
}

// =====================================================================
// NT GEMM: C[g,o] = sum_k A[g,k]*B[o,k].  128x128 tile, K-step 32,
// double-buffered LDS, async A staging, bf16 WMMA f32 accumulate.
// SWIGLU mode: B rows de-interleaved at staging (LDS rows 0..63 = glu
// rows o=2i, rows 64..127 = lin rows o=2i+1) so the gated pair lives in
// the SAME lane (acc[mt][nt] / acc[mt][nt+2]) -- no cross-lane ops.
// =====================================================================
template<int KDIM, bool SWIGLU>
__global__ __launch_bounds__(256) void magi_gemm_kernel(
    const bf16* __restrict__ A, const float* __restrict__ B,
    void* __restrict__ Out, int nbRows)
{
    const int mod  = blockIdx.z;
    const int bm   = blockIdx.y;
    const int bn   = blockIdx.x;
    const int tid  = threadIdx.x;
    const int lane = tid & 31;
    const int wave = tid >> 5;
    const int wm   = wave & 3;    // M sub-tile (32 rows each)
    const int wn   = wave >> 2;   // N sub-tile
    const int r16  = lane & 15;
    const int kg   = lane >> 4;

    __shared__ bf16 As[2][128 * 32];   // 2 x 8 KB
    __shared__ bf16 Bs[2][128 * 32];   // 2 x 8 KB

    const bf16*  Abase = A + (size_t)mod * G_TOK * KDIM + (size_t)(bm * 128) * KDIM;
    const float* Bbase = B + (size_t)mod * nbRows * KDIM + (size_t)(bn * 128) * KDIM;

    v8f acc[2][4] = {};

    const int ldRow = tid >> 1;           // 0..127
    const int ldCol = (tid & 1) * 16;     // 16 elements per thread
    int gRowB;                            // B global row for this thread's LDS row
    if constexpr (SWIGLU) gRowB = (ldRow < 64) ? (2 * ldRow) : (2 * (ldRow - 64) + 1);
    else                  gRowB = ldRow;

    float4 pb0, pb1, pb2, pb3;            // B prefetch registers (f32)

    auto stageA = [&](int k0, int buf) {
        const bf16* g = Abase + (size_t)ldRow * KDIM + k0 + ldCol;
#if HAVE_ASYNC_LDS
        lds_async_b128(g,     &As[buf][ldRow * 32 + ldCol]);
        lds_async_b128(g + 8, &As[buf][ldRow * 32 + ldCol + 8]);
#else
        uint4 a0 = ((const uint4*)g)[0];
        uint4 a1 = ((const uint4*)g)[1];
        *(uint4*)&As[buf][ldRow * 32 + ldCol]     = a0;
        *(uint4*)&As[buf][ldRow * 32 + ldCol + 8] = a1;
#endif
    };
    auto loadB = [&](int k0) {
        const float4* gb = (const float4*)(Bbase + (size_t)gRowB * KDIM + k0 + ldCol);
        pb0 = gb[0]; pb1 = gb[1]; pb2 = gb[2]; pb3 = gb[3];
    };
    auto storeB = [&](int buf) {
        bf16 tb[16];
        tb[0]=(bf16)pb0.x; tb[1]=(bf16)pb0.y; tb[2]=(bf16)pb0.z; tb[3]=(bf16)pb0.w;
        tb[4]=(bf16)pb1.x; tb[5]=(bf16)pb1.y; tb[6]=(bf16)pb1.z; tb[7]=(bf16)pb1.w;
        tb[8]=(bf16)pb2.x; tb[9]=(bf16)pb2.y; tb[10]=(bf16)pb2.z; tb[11]=(bf16)pb2.w;
        tb[12]=(bf16)pb3.x; tb[13]=(bf16)pb3.y; tb[14]=(bf16)pb3.z; tb[15]=(bf16)pb3.w;
        *(uint4*)&Bs[buf][ldRow * 32 + ldCol]     = *(const uint4*)&tb[0];
        *(uint4*)&Bs[buf][ldRow * 32 + ldCol + 8] = *(const uint4*)&tb[8];
    };
    auto compute = [&](int buf) {
        v16bf afrag[2];
        #pragma unroll
        for (int mt = 0; mt < 2; ++mt) {
            // A lane<16: K {0..7,16..23}; lane>=16: {8..15,24..31}  (ISA 7.12.2)
            const uint4* p = (const uint4*)&As[buf][(wm * 32 + mt * 16 + r16) * 32];
            ((uint4*)&afrag[mt])[0] = p[kg];
            ((uint4*)&afrag[mt])[1] = p[2 + kg];
        }
        v16bf bfrag[4];
        #pragma unroll
        for (int nt = 0; nt < 4; ++nt) {
            int br;
            if constexpr (SWIGLU)
                br = (nt < 2) ? (wn * 32 + nt * 16) : (64 + wn * 32 + (nt - 2) * 16);
            else
                br = wn * 64 + nt * 16;
            // B lane<16: K {0..7,16..23}; lane>=16: {8..15,24..31}
            const uint4* p = (const uint4*)&Bs[buf][(br + r16) * 32];
            ((uint4*)&bfrag[nt])[0] = p[kg];
            ((uint4*)&bfrag[nt])[1] = p[2 + kg];
        }
        #pragma unroll
        for (int mt = 0; mt < 2; ++mt)
            #pragma unroll
            for (int nt = 0; nt < 4; ++nt)
                acc[mt][nt] = __builtin_amdgcn_wmma_f32_16x16x32_bf16(
                    false, afrag[mt], false, bfrag[nt],
                    (short)0, acc[mt][nt], false, false);
    };

    // ---- 2-stage software pipeline: one barrier per K-step ----
    constexpr int NIT = KDIM / 32;
    stageA(0, 0);
    loadB(0);
    storeB(0);
    wait_async0();
    __syncthreads();

    for (int it = 0; it < NIT; ++it) {
        const int  cur = it & 1, nxt = cur ^ 1;
        const bool hn  = (it + 1 < NIT);
        if (hn) { stageA((it + 1) * 32, nxt); loadB((it + 1) * 32); }
        compute(cur);
        if (hn) storeB(nxt);
        wait_async0();
        __syncthreads();
    }

    // ---- epilogue (pure per-lane, full EXEC) ----
    const int mhalf = (lane >> 4) * 8;   // C layout: M = vgpr + 8*(lane/16)
    if constexpr (SWIGLU) {
        bf16* out = (bf16*)Out + (size_t)mod * G_TOK * INTER_D;
        #pragma unroll
        for (int mt = 0; mt < 2; ++mt)
            #pragma unroll
            for (int nt = 0; nt < 2; ++nt)
                #pragma unroll
                for (int e = 0; e < 8; ++e) {
                    float glu = fminf(acc[mt][nt][e], LIMIT_V);
                    float lin = fminf(fmaxf(acc[mt][nt + 2][e], -LIMIT_V), LIMIT_V);
                    float sig = __builtin_amdgcn_rcpf(1.0f + __expf(-ALPHA_V * glu));
                    float act = glu * sig * (lin + 1.0f);
                    int row = bm * 128 + wm * 32 + mt * 16 + mhalf + e;
                    int col = bn * 64 + wn * 32 + nt * 16 + r16;
                    out[(size_t)row * INTER_D + col] = (bf16)act;
                }
    } else {
        float* out = (float*)Out + (size_t)mod * G_TOK * H_DIM;
        #pragma unroll
        for (int mt = 0; mt < 2; ++mt)
            #pragma unroll
            for (int nt = 0; nt < 4; ++nt)
                #pragma unroll
                for (int e = 0; e < 8; ++e) {
                    int row = bm * 128 + wm * 32 + mt * 16 + mhalf + e;
                    int col = bn * 128 + wn * 64 + nt * 16 + r16;
                    out[(size_t)row * H_DIM + col] = acc[mt][nt][e];
                }
    }
}

// =====================================================================
extern "C" void kernel_launch(void* const* d_in, const int* in_sizes, int n_in,
                              void* d_out, int out_size, void* d_ws, size_t ws_size,
                              hipStream_t stream) {
    const float* x      = (const float*)d_in[0];
    // d_in[1] = modality_mapping: unused — tokens are pre-grouped into
    // balanced contiguous halves per the reference.
    const float* w_norm = (const float*)d_in[2];
    const float* w_up   = (const float*)d_in[3];
    const float* w_down = (const float*)d_in[4];
    float* out = (float*)d_out;

    // workspace: tm bf16 [N,H] (32 MB) | a bf16 [N,INTER] (128 MB)
    bf16* tm = (bf16*)d_ws;
    bf16* a  = (bf16*)((char*)d_ws + (size_t)N_TOK * H_DIM * sizeof(bf16));

    magi_rmsnorm_kernel<<<N_TOK, 256, 0, stream>>>(x, w_norm, tm);

    dim3 g1(UPOUT / 128, G_TOK / 128, M_MOD);      // 128 x 32 x 2
    magi_gemm_kernel<H_DIM, true><<<g1, 256, 0, stream>>>(tm, w_up, (void*)a, UPOUT);

    dim3 g2(H_DIM / 128, G_TOK / 128, M_MOD);      // 16 x 32 x 2
    magi_gemm_kernel<INTER_D, false><<<g2, 256, 0, stream>>>(a, w_down, (void*)out, H_DIM);
}
